// Decoder_43834436223379
// MI455X (gfx1250) — compile-verified, matrix operand order checked by reading, not run on hardware
//
#include <hip/hip_runtime.h>
#include <hip/hip_bf16.h>

typedef __attribute__((ext_vector_type(16))) __bf16 v16bf;
typedef __attribute__((ext_vector_type(8)))  __bf16 v8bf;
typedef __attribute__((ext_vector_type(8)))  float  v8f;

#define B_  32
#define T_  64
#define TS_ 63      // decode steps (T-1)
#define V_  32000
#define E_  256
#define H_  512
#define G_  1536    // 3*H
#define M_  2016    // B_*TS_

// ---------- helpers ----------

__device__ __forceinline__ __bf16 f32_to_bf16(float f) {
    unsigned int u = __float_as_uint(f);
    unsigned int r = (u + 0x7FFFu + ((u >> 16) & 1u)) >> 16;   // RNE
    unsigned short s = (unsigned short)r;
    __bf16 b; __builtin_memcpy(&b, &s, 2);
    return b;
}

__device__ __forceinline__ float fsigmoid(float x) {
    return 1.0f / (1.0f + __expf(-x));
}

// Load one bf16 WMMA fragment for a 16x32 (A, row=m) or 32x16 (B, col=n via
// row-major weight row n) tile. `row_base` already points at the row/col,
// contiguous in k. Per ISA layout: lanes 0-15 hold K = {k0..k0+7, k0+16..k0+23},
// lanes 16-31 hold K = {k0+8..k0+15, k0+24..k0+31}.
__device__ __forceinline__ v16bf load_frag(const __bf16* __restrict__ row_base,
                                           int k0, int half) {
    const __bf16* p = row_base + k0 + half * 8;
    v8bf lo = *(const v8bf*)(p);        // 16B load
    v8bf hi = *(const v8bf*)(p + 16);   // 16B load
    v16bf f;
#pragma unroll
    for (int i = 0; i < 8; ++i) { f[i] = lo[i]; f[i + 8] = hi[i]; }
    return f;
}

__device__ __forceinline__ v8f wmma_bf16(v16bf a, v16bf b, v8f c) {
    return __builtin_amdgcn_wmma_f32_16x16x32_bf16(false, a, false, b,
                                                   (short)0, c, false, false);
}

// ---------- one-time prep ----------

__global__ void convert_weights_kernel(const float* __restrict__ Wp,
                                       const float* __restrict__ Whh,
                                       const float* __restrict__ Wih,
                                       __bf16* __restrict__ Wp_b,
                                       __bf16* __restrict__ Whh_b,
                                       __bf16* __restrict__ Wih_b) {
    const int NP = V_ * H_, NH = G_ * H_, NI = G_ * E_;
    const int total = NP + NH + NI;
    const int stride = gridDim.x * blockDim.x;
    for (int i = blockIdx.x * blockDim.x + threadIdx.x; i < total; i += stride) {
        if (i < NP)           Wp_b[i]            = f32_to_bf16(Wp[i]);
        else if (i < NP + NH) Whh_b[i - NP]      = f32_to_bf16(Whh[i - NP]);
        else                  Wih_b[i - NP - NH] = f32_to_bf16(Wih[i - NP - NH]);
    }
}

__global__ void init_kernel(const float* __restrict__ h0,
                            float* __restrict__ out,
                            float* __restrict__ h_f32,
                            __bf16* __restrict__ h_bf) {
    const int stride = gridDim.x * blockDim.x;
    const int tid = blockIdx.x * blockDim.x + threadIdx.x;
    for (int i = tid; i < B_ * V_; i += stride) {          // out[:,0,:] = 0
        int b = i / V_, n = i - b * V_;
        out[(size_t)b * T_ * V_ + n] = 0.0f;
    }
    for (int i = tid; i < B_ * H_; i += stride) {          // h <- h0
        float v = h0[i];
        h_f32[i] = v;
        h_bf[i]  = f32_to_bf16(v);
    }
}

__global__ void gather_kernel(const int* __restrict__ trg,
                              const float* __restrict__ table,
                              __bf16* __restrict__ emb_b) {
    const int stride = gridDim.x * blockDim.x;
    for (int i = blockIdx.x * blockDim.x + threadIdx.x; i < M_ * E_; i += stride) {
        int row = i >> 8, e = i & (E_ - 1);
        int b = row / TS_, tt = row - b * TS_;             // tokens trg[:, :-1]
        int tok = trg[b * T_ + tt];
        emb_b[i] = f32_to_bf16(table[(size_t)tok * E_ + e]);
    }
}

// x_gates = emb @ W_ih^T + b_ih   : (2016 x 256) x (256 x 1536)
__global__ __launch_bounds__(256) void xgates_kernel(const __bf16* __restrict__ emb_b,
                                                     const __bf16* __restrict__ Wih_b,
                                                     const float* __restrict__ b_ih,
                                                     float* __restrict__ xg) {
    const int lane = threadIdx.x & 31;
    const int wave = blockIdx.x * 8 + (threadIdx.x >> 5);  // 0..12095, one tile each
    const int half = lane >> 4, idx = lane & 15;
    const int mt = wave / (G_ / 16), nt = wave - mt * (G_ / 16);
    const int m0 = mt * 16, n0 = nt * 16;

    const __bf16* arow = emb_b + (size_t)(m0 + idx) * E_;
    const __bf16* brow = Wih_b + (size_t)(n0 + idx) * E_;
    v8f acc = {};
    for (int k0 = 0; k0 < E_; k0 += 32) {
        v16bf a = load_frag(arow, k0, half);
        v16bf b = load_frag(brow, k0, half);
        acc = wmma_bf16(a, b, acc);
    }
    const int n = n0 + idx;
    const float bias = b_ih[n];
#pragma unroll
    for (int r = 0; r < 8; ++r) {
        int m = m0 + r + half * 8;
        xg[(size_t)m * G_ + n] = acc[r] + bias;
    }
}

// ---------- per-step kernels ----------

// Fused GRU cell: wave owns one (m-tile, j-tile); r/z/n gate tiles are three
// WMMA accumulators sharing the A fragment; gate math done in-register.
__global__ __launch_bounds__(128) void gru_cell_kernel(const __bf16* __restrict__ h_bf_prev,
                                                       const float* __restrict__ h_f32_prev,
                                                       const __bf16* __restrict__ Whh_b,
                                                       const float* __restrict__ b_hh,
                                                       const float* __restrict__ xg,
                                                       int t,
                                                       float* __restrict__ h_f32_next,
                                                       __bf16* __restrict__ h_bf_next) {
    const int lane = threadIdx.x & 31;
    const int wave = blockIdx.x * 4 + (threadIdx.x >> 5); // 0..63
    const int half = lane >> 4, idx = lane & 15;
    const int mt = wave & 1, jt = wave >> 1;              // jt in [0,32)
    const int m0 = mt * 16, j0 = jt * 16;

    const __bf16* arow   = h_bf_prev + (size_t)(m0 + idx) * H_;
    const __bf16* brow_r = Whh_b + (size_t)(j0 + idx) * H_;
    const __bf16* brow_z = Whh_b + (size_t)(H_ + j0 + idx) * H_;
    const __bf16* brow_n = Whh_b + (size_t)(2 * H_ + j0 + idx) * H_;

    v8f ar = {}, az = {}, an = {};
    for (int k0 = 0; k0 < H_; k0 += 32) {
        v16bf a  = load_frag(arow,   k0, half);
        v16bf br = load_frag(brow_r, k0, half);
        v16bf bz = load_frag(brow_z, k0, half);
        v16bf bn = load_frag(brow_n, k0, half);
        ar = wmma_bf16(a, br, ar);
        az = wmma_bf16(a, bz, az);
        an = wmma_bf16(a, bn, an);
    }

    const int j = j0 + idx;
    const float bhr = b_hh[j], bhz = b_hh[H_ + j], bhn = b_hh[2 * H_ + j];
#pragma unroll
    for (int r = 0; r < 8; ++r) {
        const int m = m0 + r + half * 8;                  // batch row
        const float* xrow = xg + ((size_t)m * TS_ + t) * G_;
        float rg = fsigmoid(xrow[j]          + ar[r] + bhr);
        float zg = fsigmoid(xrow[H_ + j]     + az[r] + bhz);
        float ng = tanhf(   xrow[2 * H_ + j] + rg * (an[r] + bhn));
        float hp = h_f32_prev[m * H_ + j];
        float hn = (1.0f - zg) * ng + zg * hp;
        h_f32_next[m * H_ + j] = hn;
        h_bf_next[m * H_ + j]  = f32_to_bf16(hn);
    }
}

// logits[b][n] = h_new[b,:] . W_proj[n,:] + b_proj[n]  -> out[b][t+1][n]
__global__ __launch_bounds__(256) void proj_kernel(const __bf16* __restrict__ h_bf,
                                                   const __bf16* __restrict__ Wp_b,
                                                   const float* __restrict__ b_proj,
                                                   float* __restrict__ out,
                                                   int t) {
    const int lane = threadIdx.x & 31;
    const int wave = blockIdx.x * 8 + (threadIdx.x >> 5); // 0..1999 N-tiles
    const int half = lane >> 4, idx = lane & 15;
    const int n0 = wave * 16;

    const __bf16* a0row = h_bf + (size_t)(idx) * H_;
    const __bf16* a1row = h_bf + (size_t)(16 + idx) * H_;
    const __bf16* brow  = Wp_b + (size_t)(n0 + idx) * H_;

    v8f acc0 = {}, acc1 = {};
    for (int k0 = 0; k0 < H_; k0 += 32) {
        v16bf b  = load_frag(brow,  k0, half);
        v16bf a0 = load_frag(a0row, k0, half);
        v16bf a1 = load_frag(a1row, k0, half);
        acc0 = wmma_bf16(a0, b, acc0);
        acc1 = wmma_bf16(a1, b, acc1);
    }

    const int n = n0 + idx;
    const float bp = b_proj[n];
#pragma unroll
    for (int r = 0; r < 8; ++r) {
        const int mA = r + half * 8;
        const int mB = 16 + mA;
        out[((size_t)mA * T_ + (t + 1)) * V_ + n] = acc0[r] + bp;
        out[((size_t)mB * T_ + (t + 1)) * V_ + n] = acc1[r] + bp;
    }
}

// Online max / sum-exp over one batch row of 32000 logits -> logsumexp.
__global__ __launch_bounds__(256) void row_lse_kernel(const float* __restrict__ out,
                                                      int t,
                                                      float* __restrict__ stats) {
    const int b = blockIdx.x;
    const float* row = out + ((size_t)b * T_ + (t + 1)) * V_;
    float m = -3.4e38f, s = 0.0f;
    for (int n = threadIdx.x; n < V_; n += 256) {
        float x = row[n];
        float mn = fmaxf(m, x);
        s = s * __expf(m - mn) + __expf(x - mn);
        m = mn;
    }
    __shared__ float sm[256], ss[256];
    sm[threadIdx.x] = m; ss[threadIdx.x] = s;
    __syncthreads();
    for (int off = 128; off > 0; off >>= 1) {
        if (threadIdx.x < off) {
            float m2 = sm[threadIdx.x + off], s2 = ss[threadIdx.x + off];
            float mn = fmaxf(sm[threadIdx.x], m2);
            ss[threadIdx.x] = ss[threadIdx.x] * __expf(sm[threadIdx.x] - mn)
                            + s2 * __expf(m2 - mn);
            sm[threadIdx.x] = mn;
        }
        __syncthreads();
    }
    if (threadIdx.x == 0) stats[b] = sm[0] + __logf(ss[0]);
}

__global__ void softmax_apply_kernel(float* __restrict__ out,
                                     const float* __restrict__ stats,
                                     int t) {
    const int stride = gridDim.x * blockDim.x;
    for (int i = blockIdx.x * blockDim.x + threadIdx.x; i < B_ * V_; i += stride) {
        int b = i / V_, n = i - b * V_;
        size_t o = ((size_t)b * T_ + (t + 1)) * V_ + n;
        out[o] = out[o] - stats[b];
    }
}

// ---------- host launcher ----------

extern "C" void kernel_launch(void* const* d_in, const int* in_sizes, int n_in,
                              void* d_out, int out_size, void* d_ws, size_t ws_size,
                              hipStream_t stream) {
    const int*   trg    = (const int*)  d_in[0];
    const float* h0     = (const float*)d_in[1];
    const float* table  = (const float*)d_in[2];
    const float* W_ih   = (const float*)d_in[3];
    const float* W_hh   = (const float*)d_in[4];
    const float* b_ih   = (const float*)d_in[5];
    const float* b_hh   = (const float*)d_in[6];
    const float* W_proj = (const float*)d_in[7];
    const float* b_proj = (const float*)d_in[8];
    float* out = (float*)d_out;

    char* ws = (char*)d_ws;
    __bf16* Wp_b  = (__bf16*)ws; ws += (size_t)V_ * H_ * 2;      // 32.8 MB
    __bf16* Whh_b = (__bf16*)ws; ws += (size_t)G_ * H_ * 2;      // 1.5 MB
    __bf16* Wih_b = (__bf16*)ws; ws += (size_t)G_ * E_ * 2;      // 0.8 MB
    __bf16* emb_b = (__bf16*)ws; ws += (size_t)M_ * E_ * 2;      // 1.0 MB
    float*  xg    = (float*) ws; ws += (size_t)M_ * G_ * 4;      // 12.4 MB
    float*  hf0   = (float*) ws; ws += (size_t)B_ * H_ * 4;
    float*  hf1   = (float*) ws; ws += (size_t)B_ * H_ * 4;
    __bf16* hb0   = (__bf16*)ws; ws += (size_t)B_ * H_ * 2;
    __bf16* hb1   = (__bf16*)ws; ws += (size_t)B_ * H_ * 2;
    float*  stats = (float*) ws; ws += 256;

    float*  hf[2] = {hf0, hf1};
    __bf16* hb[2] = {hb0, hb1};

    convert_weights_kernel<<<2048, 256, 0, stream>>>(W_proj, W_hh, W_ih,
                                                     Wp_b, Whh_b, Wih_b);
    init_kernel<<<1024, 256, 0, stream>>>(h0, out, hf[0], hb[0]);
    gather_kernel<<<1024, 256, 0, stream>>>(trg, table, emb_b);
    xgates_kernel<<<1512, 256, 0, stream>>>(emb_b, Wih_b, b_ih, xg);

    for (int t = 0; t < TS_; ++t) {
        const int cur = t & 1, nxt = cur ^ 1;
        gru_cell_kernel<<<16, 128, 0, stream>>>(hb[cur], hf[cur], Whh_b, b_hh,
                                                xg, t, hf[nxt], hb[nxt]);
        proj_kernel<<<250, 256, 0, stream>>>(hb[nxt], Wp_b, b_proj, out, t);
        row_lse_kernel<<<32, 256, 0, stream>>>(out, t, stats);
        softmax_apply_kernel<<<512, 256, 0, stream>>>(out, stats, t);
    }
}
